// VQVAEModule_43319040148082
// MI455X (gfx1250) — matrix operand order repaired and checked by reading, not compiled, not run on hardware
//
#include <hip/hip_runtime.h>

// ---------------------------------------------------------------------------
// MI455X (gfx1250) VQ-VAE forward.
// All convs -> implicit GEMM on v_wmma_f32_16x16x32_bf16 (wave32, fp32 acc).
// VQ nearest-code search -> WMMA GEMM vs LDS-resident bf16 codebook.
// Round 3: run-level bounds hoisting (K%8==0 for every conv here), b128
// weight loads, unsigned 32-bit gather offsets (saddr+voffset addressing).
// ---------------------------------------------------------------------------

typedef __attribute__((ext_vector_type(16))) __bf16 v16bf;
typedef __attribute__((ext_vector_type(8)))  float  v8f;

union Frag { uint4 u[2]; v16bf v; };

__device__ __forceinline__ __bf16 f2bf(float f) { return (__bf16)f; }

__device__ __forceinline__ v8f wmma_bf16(v16bf a, v16bf b, v8f c) {
  // (neg_a, A, neg_b, B, c_mod, C, reuse_a, reuse_b)
  return __builtin_amdgcn_wmma_f32_16x16x32_bf16(false, a, false, b, (short)0, c,
                                                 false, false);
}

// pack 8 fp32 -> 8 bf16 (16 bytes) for one ds_store_b128
__device__ __forceinline__ uint4 pack8bf(const float* v) {
  __bf16 t[8];
  #pragma unroll
  for (int j = 0; j < 8; ++j) t[j] = (__bf16)v[j];
  uint4 u;
  __builtin_memcpy(&u, t, 16);
  return u;
}

// ---------------------------------------------------------------------------
// Generic implicit-GEMM conv: out[b,co,oh,ow] = sum_{ci,kh,kw} in*w (+bias,+res)
// Supports stride, padding, lhs_dilation (transposed conv), relu-on-input.
// Block = 256 threads = 8 waves. Block tile: 64 out-channels x 64 positions.
// Wave (mi, nhalf) computes two 16x16 WMMA tiles sharing one A fragment.
// ---------------------------------------------------------------------------
#define KS  32
#define MT  64
#define NTL 64
#define SAS 40   // LDS row stride (bf16): 80B = 5*16B -> aligned, conflict-free
#define SBS 40

__global__ __launch_bounds__(256)
void conv_implicit_wmma(const float* __restrict__ in, const float* __restrict__ w,
                        const float* __restrict__ bias, const float* __restrict__ resid,
                        float* __restrict__ out,
                        int B, int Cin, int Hin, int Win,
                        int Cout, int Kh, int Kw,
                        int stride, int pad, int dshift,
                        int Hout, int Wout, int in_relu,
                        unsigned mKhKw, unsigned mKw, int khkw)
{
  __shared__ __align__(16) __bf16 sA[MT  * SAS];   // weight slab [M][K]
  __shared__ __align__(16) __bf16 sB[NTL * SBS];   // patch  slab [pos][K]

  const int tid   = threadIdx.x;
  const int wave  = tid >> 5;
  const int lane  = tid & 31;
  const int mi    = wave >> 1;          // 0..3  -> M sub-tile
  const int nhalf = wave & 1;           // 0..1  -> pair of N sub-tiles

  const int K      = Cin * Kh * Kw;     // NOTE: K % 8 == 0 for every conv here
  const int HW     = Hout * Wout;
  const int HinWin = Hin * Win;
  const int p_blk  = blockIdx.x * NTL;
  const int m_blk  = blockIdx.y * MT;
  const int dmask  = (1 << dshift) - 1;

  // staging coordinates (invariant over K loop): 64 rows x 4 threads x 8 elems
  const int srow = tid >> 2;
  const int skk  = (tid & 3) * 8;

  // position decode for the B row this thread stages (constant over K loop)
  const int bp   = p_blk + srow;
  const int bb   = bp / HW;
  const int brem = bp - bb * HW;
  const int boh  = brem / Wout;
  const int bow  = brem - boh * Wout;
  const int bohs = boh * stride - pad;
  const int bows = bow * stride - pad;
  const int bbase = bb * Cin * HinWin;  // batch base offset into input

  const int am   = m_blk + srow;        // weight row this thread stages
  const bool mok = (am < Cout);

  v8f acc0 = {0.f, 0.f, 0.f, 0.f, 0.f, 0.f, 0.f, 0.f};
  v8f acc1 = {0.f, 0.f, 0.f, 0.f, 0.f, 0.f, 0.f, 0.f};

  // fragment read coordinates
  const int art  = mi * 16 + (lane & 15);          // A row (M)
  const int akf  = (lane >> 4) * 16;               // A K-phase
  const int brt0 = (nhalf * 2) * 16 + (lane & 15); // B row (position), tile 0
  const int brt1 = brt0 + 16;                      //                 tile 1
  const int bkf  = (lane >> 4) * 16;               // B K-phase

  const int KT = (K + KS - 1) / KS;
  for (int kt = 0; kt < KT; ++kt) {
    const int k0 = kt * KS;
    __syncthreads();

    // ---- stage A: weights (OIHW flattens to [Cout][K]) ----
    // K % 8 == 0, so an 8-run is in-bounds iff its first element is: one
    // run-level guard, two b128 loads from a clamped base, branchless zero.
    {
      const bool aok = mok & ((k0 + skk) < K);
      const unsigned abase = aok ? (unsigned)(am * K + k0 + skk) : 0u;
      const float4 f0 = *(const float4*)(w + abase);
      const float4 f1 = *(const float4*)(w + abase + 4);
      float av[8] = {f0.x, f0.y, f0.z, f0.w, f1.x, f1.y, f1.z, f1.w};
      #pragma unroll
      for (int j = 0; j < 8; ++j) av[j] = aok ? av[j] : 0.f;
      *(uint4*)&sA[srow * SAS + skk] = pack8bf(av);
    }
    // ---- stage B: gathered input patches; branchless predicated loads ----
    {
      const bool runok = (k0 + skk) < K;           // run-level K guard
      float bv[8];
      #pragma unroll
      for (int j = 0; j < 8; ++j) {
        const int k = k0 + skk + j;
        const unsigned ci = ((unsigned)k * mKhKw) >> 16;    // k / (Kh*Kw)
        const int kr = k - (int)ci * khkw;
        const unsigned kh = ((unsigned)kr * mKw) >> 16;     // kr / Kw
        const int kw = kr - (int)kh * Kw;
        const int ihe = bohs + (int)kh;
        const int iwe = bows + kw;
        const int ih  = ihe >> dshift;                      // lhs_dilation in {1,2}
        const int iw  = iwe >> dshift;
        const bool ok = runok & (ihe >= 0) & (iwe >= 0) &
                        ((ihe & dmask) == 0) & ((iwe & dmask) == 0) &
                        (ih < Hin) & (iw < Win);
        const unsigned a =
            ok ? (unsigned)(bbase + (int)ci * HinWin + ih * Win + iw) : 0u;
        float v = in[a];
        if (in_relu) v = fmaxf(v, 0.f);
        bv[j] = ok ? v : 0.f;
      }
      *(uint4*)&sB[srow * SBS + skk] = pack8bf(bv);
    }
    __syncthreads();

    // ---- fragments (wave32 WMMA VGPR layouts, ISA 7.12.2) ----
    Frag fa, fb0, fb1;
    fa.u[0]  = *(const uint4*)&sA[art  * SAS + akf];        // K = akf..akf+7
    fa.u[1]  = *(const uint4*)&sA[art  * SAS + akf + 8];    // K = akf+8..+15
    fb0.u[0] = *(const uint4*)&sB[brt0 * SBS + bkf];        // K = bkf..bkf+15
    fb0.u[1] = *(const uint4*)&sB[brt0 * SBS + bkf + 8];
    fb1.u[0] = *(const uint4*)&sB[brt1 * SBS + bkf];
    fb1.u[1] = *(const uint4*)&sB[brt1 * SBS + bkf + 8];

    acc0 = wmma_bf16(fa.v, fb0.v, acc0);
    acc1 = wmma_bf16(fa.v, fb1.v, acc1);
  }

  // ---- epilogue: bias + residual add, scatter to NCHW fp32 ----
  const int mw = m_blk + mi * 16 + (lane >> 4) * 8;  // C/D: row M = r + 8*(lane>>4)
  #pragma unroll
  for (int t = 0; t < 2; ++t) {
    const v8f acc = t ? acc1 : acc0;
    const int n   = p_blk + (nhalf * 2 + t) * 16 + (lane & 15);
    const int b   = n / HW;
    const int rem = n - b * HW;
    const int oh  = rem / Wout;
    const int ow  = rem - oh * Wout;
    #pragma unroll
    for (int r = 0; r < 8; ++r) {
      const int m = mw + r;
      if (m < Cout) {
        float v = acc[r];
        if (bias)  v += bias[m];
        const int oidx = ((b * Cout + m) * Hout + oh) * Wout + ow;
        if (resid) v += resid[oidx];
        out[oidx] = v;
      }
    }
  }
}

// ---------------------------------------------------------------------------
// VQ argmin: per position p, argmin_k ( ||c_k||^2 - 2 z_p . c_k )
// Codebook (512x64) staged to LDS as bf16; scores via WMMA.
// Block = 8 waves x 16 positions = 128 positions; grid = 512.
// ---------------------------------------------------------------------------
#define CBS 72   // padded codebook row stride (144B, 16B aligned, conflict-free)

__global__ __launch_bounds__(256)
void vq_argmin_wmma(const float* __restrict__ z, const float* __restrict__ cb,
                    int* __restrict__ idx)
{
  __shared__ __align__(16) __bf16 sC[512 * CBS];
  __shared__ float sN[512];
  const int tid  = threadIdx.x;
  const int wave = tid >> 5;
  const int lane = tid & 31;

  for (int r = tid; r < 512; r += 256) {
    float s = 0.f;
    #pragma unroll
    for (int d8 = 0; d8 < 8; ++d8) {
      const float4 f0 = *(const float4*)&cb[r * 64 + d8 * 8];
      const float4 f1 = *(const float4*)&cb[r * 64 + d8 * 8 + 4];
      s += f0.x * f0.x + f0.y * f0.y + f0.z * f0.z + f0.w * f0.w;
      s += f1.x * f1.x + f1.y * f1.y + f1.z * f1.z + f1.w * f1.w;
      float v[8] = {f0.x, f0.y, f0.z, f0.w, f1.x, f1.y, f1.z, f1.w};
      *(uint4*)&sC[r * CBS + d8 * 8] = pack8bf(v);
    }
    sN[r] = s;
  }
  __syncthreads();

  // A fragments: 16 positions x 64 features (two K=32 fragments), loaded once.
  const int p0 = blockIdx.x * 128 + wave * 16;
  const int p  = p0 + (lane & 15);
  const int pb = p >> 10;          // batch (H*W = 1024)
  const int hw = p & 1023;
  const int kb = (lane >> 4) * 16;

  __bf16 ta0[16], ta1[16];
  #pragma unroll
  for (int e = 0; e < 16; ++e) {
    const int d = kb + e;                        // A-matrix 16x32 element map (contiguous k)
    ta0[e] = f2bf(z[(unsigned)(pb * 65536 + d * 1024 + hw)]);
    ta1[e] = f2bf(z[(unsigned)(pb * 65536 + (d + 32) * 1024 + hw)]);
  }
  v16bf a0, a1;
  __builtin_memcpy(&a0, ta0, 32);
  __builtin_memcpy(&a1, ta1, 32);

  float best[8];
  int   bidx[8];
  #pragma unroll
  for (int r = 0; r < 8; ++r) { best[r] = 3.4e38f; bidx[r] = 0; }

  for (int ct = 0; ct < 32; ++ct) {
    const int code = ct * 16 + (lane & 15);      // B: lane&15 = col N = code
    const int bk   = (lane >> 4) * 16;
    Frag b0, b1;
    b0.u[0] = *(const uint4*)&sC[code * CBS + bk];
    b0.u[1] = *(const uint4*)&sC[code * CBS + bk + 8];
    b1.u[0] = *(const uint4*)&sC[code * CBS + 32 + bk];
    b1.u[1] = *(const uint4*)&sC[code * CBS + 32 + bk + 8];
    v8f acc = {0.f, 0.f, 0.f, 0.f, 0.f, 0.f, 0.f, 0.f};
    acc = wmma_bf16(a0, b0.v, acc);
    acc = wmma_bf16(a1, b1.v, acc);
    const float cn = sN[code];
    #pragma unroll
    for (int r = 0; r < 8; ++r) {
      const float dist = cn - 2.f * acc[r];
      if (dist < best[r]) { best[r] = dist; bidx[r] = code; }
    }
  }

  // min/argmin across the 16 lanes holding the same rows (wave32 shuffles)
  #pragma unroll
  for (int r = 0; r < 8; ++r) {
    float d = best[r]; int i = bidx[r];
    #pragma unroll
    for (int off = 1; off < 16; off <<= 1) {
      const float od = __shfl_xor(d, off, 32);
      const int   oi = __shfl_xor(i, off, 32);
      if (od < d || (od == d && oi < i)) { d = od; i = oi; }
    }
    if ((lane & 15) == 0) idx[p0 + (lane >> 4) * 8 + r] = i;
  }
}

// ---------------------------------------------------------------------------
// q = codebook[idx] scattered back to NCHW; accumulate sum((q-z)^2).
// ---------------------------------------------------------------------------
__global__ __launch_bounds__(256)
void vq_gather_stats(const float* __restrict__ z, const float* __restrict__ cb,
                     const int* __restrict__ idx, float* __restrict__ q,
                     float* __restrict__ msesum)
{
  const int t = blockIdx.x * 256 + threadIdx.x;     // 4.19M threads
  const int d = t >> 16;
  const int p = t & 65535;
  const int code = idx[p];
  const int b = p >> 10, hw = p & 1023;
  const unsigned a = (unsigned)(b * 65536 + d * 1024 + hw);
  const float v = cb[(unsigned)(code * 64 + d)];
  const float diff = v - z[a];
  q[a] = v;
  __shared__ float red[256];
  red[threadIdx.x] = diff * diff;
  __syncthreads();
  for (int s = 128; s; s >>= 1) {
    if (threadIdx.x < s) red[threadIdx.x] += red[threadIdx.x + s];
    __syncthreads();
  }
  if (threadIdx.x == 0) atomicAdd(msesum, red[0]);
}

__global__ __launch_bounds__(256)
void vq_hist(const int* __restrict__ idx, int* __restrict__ hist)
{
  __shared__ int h[512];
  h[threadIdx.x] = 0; h[threadIdx.x + 256] = 0;
  __syncthreads();
  const int p = blockIdx.x * 256 + threadIdx.x;
  atomicAdd(&h[idx[p]], 1);
  __syncthreads();
  atomicAdd(&hist[threadIdx.x],       h[threadIdx.x]);
  atomicAdd(&hist[threadIdx.x + 256], h[threadIdx.x + 256]);
}

__global__ __launch_bounds__(256)
void vq_finalize(const float* __restrict__ msesum, const int* __restrict__ hist,
                 float* __restrict__ out2)
{
  __shared__ float red[256];
  const int t = threadIdx.x;
  float ent = 0.f;
  for (int k = t; k < 512; k += 256) {
    const float pr = (float)hist[k] * (1.0f / 65536.0f);
    ent += pr * logf(pr + 1e-10f);
  }
  red[t] = ent;
  __syncthreads();
  for (int s = 128; s; s >>= 1) {
    if (t < s) red[t] += red[t + s];
    __syncthreads();
  }
  if (t == 0) {
    out2[0] = msesum[0] * 1.25f / 4194304.0f;   // (1 + commitment 0.25) * mean
    out2[1] = expf(-red[0]);                    // perplexity
  }
}

// ---------------------------------------------------------------------------
// Host: pointer mapping + pipeline orchestration (stream-only, graph-safe).
// ---------------------------------------------------------------------------
extern "C" void kernel_launch(void* const* d_in, const int* in_sizes, int n_in,
                              void* d_out, int out_size, void* d_ws, size_t ws_size,
                              hipStream_t stream)
{
  (void)n_in; (void)out_size; (void)ws_size;
  auto G = [&](int i) { return (const float*)d_in[i]; };

  const float *X, *EW1, *EB1, *EW2, *EB2, *EW3, *EB3;
  const float *EA0, *EWB0, *EA1, *EWB1, *PW, *PB, *CB;
  const float *DW1, *DB1, *DA0, *DWB0, *DA1, *DWB1, *WT1, *BT1, *WT2, *BT2;

  if (in_sizes[0] == 64 * 3 * 128 * 128) {
    // dict-insertion order
    X = G(0);  EW1 = G(1);  EB1 = G(2);  EW2 = G(3);  EB2 = G(4);
    EW3 = G(5); EB3 = G(6);
    EA0 = G(7); EWB0 = G(8); EA1 = G(9); EWB1 = G(10);
    PW = G(11); PB = G(12); CB = G(13);
    DW1 = G(14); DB1 = G(15);
    DA0 = G(16); DWB0 = G(17); DA1 = G(18); DWB1 = G(19);
    WT1 = G(20); BT1 = G(21); WT2 = G(22); BT2 = G(23);
  } else {
    // jax pytree (alphabetically sorted keys) order
    CB = G(0);
    DB1 = G(1); BT1 = G(2); BT2 = G(3);
    DA0 = G(4); DWB0 = G(5); DA1 = G(6); DWB1 = G(7);
    DW1 = G(8); WT1 = G(9); WT2 = G(10);
    EB1 = G(11); EB2 = G(12); EB3 = G(13);
    EA0 = G(14); EWB0 = G(15); EA1 = G(16); EWB1 = G(17);
    EW1 = G(18); EW2 = G(19); EW3 = G(20);
    PB = G(21); PW = G(22); X = G(23);
  }

  // ---- scratch arena (fp32, bump alloc with reuse) ----
  float* ws = (float*)d_ws;
  size_t off = 0;
  auto alloc = [&](size_t n) {
    float* p = ws + off;
    off += (n + 255) & ~(size_t)255;
    return p;
  };
  float* h1 = alloc(16777216);   // [64,64,64,64]
  float* h2 = alloc(8388608);    // [64,128,32,32]
  float* x3 = alloc(8388608);
  float* x4 = alloc(8388608);
  float* rh = alloc(2097152);    // [64,32,32,32]
  float* z  = alloc(4194304);    // [64,64,32,32]
  float* q  = alloc(4194304);
  int*   idxb  = (int*)alloc(65536);
  float* stats = alloc(1 + 512); // [0]=mse sum, [1..512]=hist (ints)
  int*   hist  = (int*)(stats + 1);

  hipMemsetAsync(stats, 0, (1 + 512) * sizeof(float), stream);

  auto conv = [&](const float* in, const float* w, const float* bias,
                  const float* resid, float* out,
                  int B, int Cin, int Hin, int Win, int Cout, int Kh, int Kw,
                  int st, int pd, int dl, int Ho, int Wo, int in_relu) {
    dim3 grid((unsigned)((B * Ho * Wo) / NTL), (unsigned)((Cout + MT - 1) / MT));
    const int khkw = Kh * Kw;
    const unsigned mKhKw = (65536u + khkw - 1) / khkw;  // ceil(2^16/d) magic
    const unsigned mKw   = (65536u + Kw - 1) / Kw;
    const int dshift = (dl == 2) ? 1 : 0;
    conv_implicit_wmma<<<grid, 256, 0, stream>>>(in, w, bias, resid, out,
                                                 B, Cin, Hin, Win, Cout, Kh, Kw,
                                                 st, pd, dshift, Ho, Wo, in_relu,
                                                 mKhKw, mKw, khkw);
  };

  // ---- encoder ----
  conv(X,  EW1, EB1, nullptr, h1, 64,   3, 128, 128,  64, 4, 4, 2, 1, 1,  64,  64, 0);
  conv(h1, EW2, EB2, nullptr, h2, 64,  64,  64,  64, 128, 4, 4, 2, 1, 1,  32,  32, 1);
  conv(h2, EW3, EB3, nullptr, x3, 64, 128,  32,  32, 128, 3, 3, 1, 1, 1,  32,  32, 1);
  conv(x3, EA0,  nullptr, nullptr, rh, 64, 128, 32, 32,  32, 3, 3, 1, 1, 1, 32, 32, 1);
  conv(rh, EWB0, nullptr, x3,      x4, 64,  32, 32, 32, 128, 1, 1, 1, 0, 1, 32, 32, 1);
  conv(x4, EA1,  nullptr, nullptr, rh, 64, 128, 32, 32,  32, 3, 3, 1, 1, 1, 32, 32, 1);
  conv(rh, EWB1, nullptr, x4,      x3, 64,  32, 32, 32, 128, 1, 1, 1, 0, 1, 32, 32, 1);
  conv(x3, PW, PB, nullptr, z, 64, 128, 32, 32, 64, 1, 1, 1, 0, 1, 32, 32, 1);

  // ---- vector quantization ----
  vq_argmin_wmma<<<512, 256, 0, stream>>>(z, CB, idxb);
  vq_gather_stats<<<16384, 256, 0, stream>>>(z, CB, idxb, q, stats);
  vq_hist<<<256, 256, 0, stream>>>(idxb, hist);

  // ---- decoder (straight-through forward value == q) ----
  conv(q,  DW1, DB1, nullptr, h2, 64,  64, 32, 32, 128, 3, 3, 1, 1, 1, 32, 32, 0);
  conv(h2, DA0,  nullptr, nullptr, rh, 64, 128, 32, 32,  32, 3, 3, 1, 1, 1, 32, 32, 1);
  conv(rh, DWB0, nullptr, h2,      x3, 64,  32, 32, 32, 128, 1, 1, 1, 0, 1, 32, 32, 1);
  conv(x3, DA1,  nullptr, nullptr, rh, 64, 128, 32, 32,  32, 3, 3, 1, 1, 1, 32, 32, 1);
  conv(rh, DWB1, nullptr, x3,      x4, 64,  32, 32, 32, 128, 1, 1, 1, 0, 1, 32, 32, 1);
  // ConvTranspose2d(k=4,s=2,p=1) == lhs-dilated conv, pad 2, dil 2
  conv(x4, WT1, BT1, nullptr, h1, 64, 128, 32, 32,  64, 4, 4, 1, 2, 2,  64,  64, 1);
  conv(h1, WT2, BT2, nullptr, (float*)d_out, 64, 64, 64, 64, 3, 4, 4, 1, 2, 2, 128, 128, 1);

  // ---- scalars: loss, perplexity ----
  vq_finalize<<<1, 256, 0, stream>>>(stats, hist, (float*)d_out + 3145728);
}
